// PAM_1391569404258
// MI455X (gfx1250) — matrix operand order repaired and checked by reading, not compiled, not run on hardware
//
#include <hip/hip_runtime.h>
#include <hip/hip_bf16.h>
#include <math.h>

typedef __attribute__((ext_vector_type(16))) _Float16 v16h;
typedef __attribute__((ext_vector_type(8)))  _Float16 v8h;
typedef __attribute__((ext_vector_type(8)))  float    v8f;

#define NBATCH 4
#define HW     4096
#define CC     256
#define CQ     32

__device__ __forceinline__ v16h cat8(v8h lo, v8h hi) {
    v16h r;
#pragma unroll
    for (int i = 0; i < 8; ++i) { r[i] = lo[i]; r[8 + i] = hi[i]; }
    return r;
}

// ---------------------------------------------------------------------------
// Projection GEMM: X[4096,256](f32) x W[256,LDN](f32) -> f16.
// LDN is a template constant so the 16 strided weight loads are a clause of
// independent loads with immediate offsets.
// TRANSPOSE_STORE=true  -> dst is [LDN, hw]  (V: contiguous B-fragments later)
// TRANSPOSE_STORE=false -> dst is [hw, LDN]  (Q, K row-major)
// One wave = one 16x16 tile, K-loop = 8 x V_WMMA_F32_16X16X32_F16.
// ---------------------------------------------------------------------------
template<int LDN, bool TRANSPOSE_STORE>
__global__ __launch_bounds__(128)
void pam_proj_kernel(const float* __restrict__ x,
                     const float* __restrict__ W,
                     _Float16* __restrict__ dst)
{
    constexpr int NT = LDN / 16;
    const int lane = threadIdx.x & 31;
    const int wave = threadIdx.x >> 5;
    const int tile = blockIdx.x * 4 + wave;            // batch*256*NT tiles
    const int batch = tile / (256 * NT);
    const int rem   = tile % (256 * NT);
    const int mt = rem / NT;
    const int nt = rem % NT;
    const int row0  = mt * 16;
    const int ncol0 = nt * 16;

    const int hh = lane >> 4;          // half-group (ISA frag layout tables)
    const int ln = lane & 15;
    const int kbA = hh * 8;            // A-frag: K = kbA+i (i<8), kbA+8+i (i>=8)
    const int kbB = hh * 16;           // unused; B-frag uses the same K order as A (kbA)

    const float* xrow = x + ((size_t)(batch * HW + row0 + ln)) * CC;
    const float* wcol = W + (size_t)kbA * LDN + (ncol0 + ln);

    v8f acc = {};
#pragma unroll
    for (int kc = 0; kc < CC; kc += 32) {
        const float* ap = xrow + kc + kbA;
        v16h a;
#pragma unroll
        for (int i = 0; i < 8; ++i) {
            a[i]     = (_Float16)ap[i];
            a[8 + i] = (_Float16)ap[16 + i];
        }
        const float* bp = wcol + (size_t)kc * LDN;     // constant offsets i*LDN
        v16h b;
#pragma unroll
        for (int i = 0; i < 16; ++i) b[i] = (_Float16)bp[(i < 8 ? i : i + 8) * LDN];

        acc = __builtin_amdgcn_wmma_f32_16x16x32_f16(false, a, false, b,
                                                     (short)0, acc, false, false);
    }

    if (TRANSPOSE_STORE) {
        // C-layout vgprs r=0..7 are 8 consecutive output rows (= keys) for a
        // fixed column (= channel): one packed 16B store per lane.
        const int ch = ncol0 + ln;
        v8h o;
#pragma unroll
        for (int r = 0; r < 8; ++r) o[r] = (_Float16)acc[r];
        *(v8h*)(dst + ((size_t)(batch * LDN + ch)) * HW + row0 + hh * 8) = o;
    } else {
        const int col = ncol0 + ln;
#pragma unroll
        for (int r = 0; r < 8; ++r)
            dst[((size_t)(batch * HW + row0 + r + hh * 8)) * LDN + col] =
                (_Float16)acc[r];
    }
}

// ---------------------------------------------------------------------------
// Flash attention, BN = 64 keys per chunk (no score materialization).
// Block = 128 threads = 4 waves.  Wave w: qtile = w>>1 (16 q rows),
// chalf = w&1 (channels 0-127 / 128-255).  Grid = 4 batches * 128 q-blocks.
// All streamed loads use uniform-scalar-base + invariant 32-bit lane offset
// (GVS addressing) so per-iteration address math is SALU-only, and K
// fragments are software-pipelined one chunk ahead.
// ---------------------------------------------------------------------------
__global__ __launch_bounds__(128)
void pam_attn_kernel(const _Float16* __restrict__ Qh,
                     const _Float16* __restrict__ Kh,
                     const _Float16* __restrict__ Vt,
                     const float* __restrict__ x,
                     const float* __restrict__ gamma,
                     float* __restrict__ out)
{
    __shared__ __align__(16) _Float16 pbuf[4][16 * 64];

    const int lane  = threadIdx.x & 31;
    const int wave  = threadIdx.x >> 5;
    const int qtile = wave >> 1;
    const int chalf = wave & 1;
    const int batch = blockIdx.x >> 7;
    const int qblk  = blockIdx.x & 127;
    const int q0    = qblk * 32 + qtile * 16;

    const int hh  = lane >> 4;
    const int ln  = lane & 15;
    const int kbA = hh * 8;

    // uniform bases (SGPRs) + loop-invariant per-lane offsets (VGPRs)
    const _Float16* kb = Kh + (size_t)batch * HW * CQ;
    const _Float16* vb = Vt + (size_t)batch * CC * HW;
    const int koff = ln * CQ + hh * 16;            // K-frag lane offset
    int voff[8];                                    // V-frag lane offsets
#pragma unroll
    for (int t = 0; t < 8; ++t)
        voff[t] = ((chalf * 8 + t) * 16 + ln) * HW + hh * 16;

    // persistent Q fragment (A layout: row = ln, K = 16*hh+i)
    const _Float16* qp = Qh + ((size_t)(batch * HW + q0 + ln)) * CQ + hh * 16;
    const v16h qf = cat8(*(const v8h*)qp, *(const v8h*)(qp + 8));

    const v8f zero = {};
    v8f acc[8];
#pragma unroll
    for (int t = 0; t < 8; ++t) acc[t] = zero;
    float mrow[8], lrow[8];
#pragma unroll
    for (int r = 0; r < 8; ++r) { mrow[r] = -INFINITY; lrow[r] = 0.f; }

    _Float16* pw = &pbuf[wave][0];

    // preload K fragments for the first chunk
    v16h kf[4];
#pragma unroll
    for (int nb = 0; nb < 4; ++nb) {
        const _Float16* kp = kb + koff + nb * 16 * CQ;
        kf[nb] = cat8(*(const v8h*)kp, *(const v8h*)(kp + 8));
    }

    for (int key0 = 0; key0 < HW; key0 += 64) {
        // ---- S = Q K^T : four 16x16 tiles from pipelined fragments ----
        v8f s[4];
#pragma unroll
        for (int nb = 0; nb < 4; ++nb)
            s[nb] = __builtin_amdgcn_wmma_f32_16x16x32_f16(false, qf, false, kf[nb],
                                                           (short)0, zero, false, false);

        // ---- prefetch next chunk's K fragments (wraps; latency hidden) ----
        const int key1 = (key0 + 64) & (HW - 1);
        const _Float16* kbn = kb + (size_t)key1 * CQ;
#pragma unroll
        for (int nb = 0; nb < 4; ++nb) {
            const _Float16* kp = kbn + koff + nb * 16 * CQ;
            kf[nb] = cat8(*(const v8h*)kp, *(const v8h*)(kp + 8));
        }

        // ---- online softmax (row stats per C-layout vgpr) ----
        float scale[8];
#pragma unroll
        for (int r = 0; r < 8; ++r) {
            float t = fmaxf(fmaxf(s[0][r], s[1][r]), fmaxf(s[2][r], s[3][r]));
#pragma unroll
            for (int m = 1; m < 16; m <<= 1) t = fmaxf(t, __shfl_xor(t, m, 32));
            const float mnew = fmaxf(mrow[r], t);
            scale[r] = __expf(mrow[r] - mnew);
            mrow[r] = mnew;
            float ssum = 0.f;
#pragma unroll
            for (int nb = 0; nb < 4; ++nb) {
                const float p = __expf(s[nb][r] - mnew);
                ssum += p;
                pw[(r + 8 * hh) * 64 + nb * 16 + ln] = (_Float16)p;
            }
#pragma unroll
            for (int m = 1; m < 16; m <<= 1) ssum += __shfl_xor(ssum, m, 32);
            lrow[r] = lrow[r] * scale[r] + ssum;
        }
#pragma unroll
        for (int t = 0; t < 8; ++t)
#pragma unroll
            for (int r = 0; r < 8; ++r) acc[t][r] *= scale[r];

        // ---- reload P as two A fragments (same-wave DS ops are in-order) ----
        const _Float16* pp = pw + ln * 64;
        v16h pf0 = cat8(*(const v8h*)(pp + hh * 16),      *(const v8h*)(pp + hh * 16 + 8));
        v16h pf1 = cat8(*(const v8h*)(pp + 32 + hh * 16), *(const v8h*)(pp + 32 + hh * 16 + 8));

        // ---- out += P * V over 8 channel tiles ----
        const _Float16* vbk = vb + key0;            // uniform (SALU advance)
#pragma unroll
        for (int t = 0; t < 8; ++t) {
            const _Float16* vp = vbk + voff[t];
            v16h vf0 = cat8(*(const v8h*)vp,        *(const v8h*)(vp + 8));
            v16h vf1 = cat8(*(const v8h*)(vp + 32), *(const v8h*)(vp + 40));
            acc[t] = __builtin_amdgcn_wmma_f32_16x16x32_f16(false, pf0, false, vf0,
                                                            (short)0, acc[t], false, false);
            acc[t] = __builtin_amdgcn_wmma_f32_16x16x32_f16(false, pf1, false, vf1,
                                                            (short)0, acc[t], false, false);
        }
    }

    // ---- epilogue: out = gamma * (acc / l) + x ----
    const float g = gamma[0];
    const float* xb = x + (size_t)batch * HW * CC;   // uniform bases
    float* ob = out + (size_t)batch * HW * CC;
    float inv[8];
#pragma unroll
    for (int r = 0; r < 8; ++r) inv[r] = 1.f / lrow[r];
#pragma unroll
    for (int t = 0; t < 8; ++t) {
        const int ch = (chalf * 8 + t) * 16 + ln;
#pragma unroll
        for (int r = 0; r < 8; ++r) {
            const int off = (q0 + r + 8 * hh) * CC + ch;   // 32-bit lane offset
            ob[off] = g * (acc[t][r] * inv[r]) + xb[off];
        }
    }
}

extern "C" void kernel_launch(void* const* d_in, const int* in_sizes, int n_in,
                              void* d_out, int out_size, void* d_ws, size_t ws_size,
                              hipStream_t stream) {
    const float* x     = (const float*)d_in[0];
    const float* Wb    = (const float*)d_in[1];
    const float* Wc    = (const float*)d_in[2];
    const float* Wd    = (const float*)d_in[3];
    const float* gamma = (const float*)d_in[4];

    _Float16* Qh = (_Float16*)d_ws;                       // [4, 4096, 32]
    _Float16* Kh = Qh + (size_t)NBATCH * HW * CQ;         // [4, 4096, 32]
    _Float16* Vt = Kh + (size_t)NBATCH * HW * CQ;         // [4, 256, 4096]

    // Q, K: 4*256*2 tiles each, 4 waves/block -> 512 blocks
    pam_proj_kernel<CQ, false><<<512,  128, 0, stream>>>(x, Wb, Qh);
    pam_proj_kernel<CQ, false><<<512,  128, 0, stream>>>(x, Wc, Kh);
    // V: 4*256*16 tiles, transposed store -> 4096 blocks
    pam_proj_kernel<CC, true ><<<4096, 128, 0, stream>>>(x, Wd, Vt);

    // 4 batches * 128 q-blocks of 32 rows
    pam_attn_kernel<<<512, 128, 0, stream>>>(Qh, Kh, Vt, x, gamma, (float*)d_out);
}
